// KGQueryMPNN_26233660244697
// MI455X (gfx1250) — matrix operand order repaired and checked by reading, not compiled, not run on hardware
//
#include <hip/hip_runtime.h>
#include <hip/hip_bf16.h>
#include <math.h>

typedef __attribute__((ext_vector_type(16))) __bf16 v16bf;
typedef __attribute__((ext_vector_type(8)))  __bf16 v8bf;
typedef __attribute__((ext_vector_type(8)))  float  v8f;

#define NEG_SLOPE 0.2f

// -------- helpers --------
__device__ __forceinline__ void atomicMaxF(float* addr, float v) {
    if (v >= 0.0f) atomicMax((int*)addr, __float_as_int(v));
    else           atomicMin((unsigned int*)addr, __float_as_uint(v));
}

// -------- pre-pass: self-loop attr = mean of incoming edge attrs --------
__global__ void prep_zero_kernel(float* __restrict__ loopS, float* __restrict__ deg, int N) {
    int i = blockIdx.x * blockDim.x + threadIdx.x;
    if (i < N * 128) loopS[i] = 0.0f;
    if (i < N) deg[i] = 0.0f;
}

__global__ void prep_accum_kernel(const int* __restrict__ dstA, const float* __restrict__ ea,
                                  float* __restrict__ loopS, float* __restrict__ deg, int E) {
    int gid = blockIdx.x * blockDim.x + threadIdx.x;
    int e = gid >> 5;
    if (e >= E) return;
    int q = (gid & 31) * 4;
    int d = dstA[e];
    const float4 v = *(const float4*)(ea + (size_t)e * 128 + q);
    float* o = loopS + (size_t)d * 128 + q;
    atomicAdd(o + 0, v.x);
    atomicAdd(o + 1, v.y);
    atomicAdd(o + 2, v.z);
    atomicAdd(o + 3, v.w);
    if ((gid & 31) == 0) atomicAdd(&deg[d], 1.0f);
}

__global__ void prep_final_kernel(float* __restrict__ loopS, const float* __restrict__ deg, int N) {
    int i = blockIdx.x * blockDim.x + threadIdx.x;
    if (i >= N * 128) return;
    float c = deg[i >> 7];
    c = (c > 1.0f) ? c : 1.0f;
    loopS[i] = loopS[i] / c;
}

// -------- weight packing: f32 [128x128] -> bf16 B-operand layout --------
// Packed index: ((c*4 + s)*32 + lane)*16 + i  holds  W[s*32 + (lane>=16?8:0) + (i&7) + (i>=8?16:0)][c*16 + (lane&15)]
__global__ void pack_w_kernel(const float* __restrict__ W0, const float* __restrict__ W1,
                              const float* __restrict__ W2,
                              __bf16* __restrict__ P0, __bf16* __restrict__ P1,
                              __bf16* __restrict__ P2) {
    int gid = blockIdx.x * blockDim.x + threadIdx.x;
    if (gid >= 3 * 16384) return;
    int w = gid >> 14;
    int r = gid & 16383;
    int i = r & 15;
    int l = (r >> 4) & 31;
    int s = (r >> 9) & 3;
    int c = r >> 11;
    const float* W = (w == 0) ? W0 : ((w == 1) ? W1 : W2);
    __bf16* P = (w == 0) ? P0 : ((w == 1) ? P1 : P2);
    int k = s * 32 + ((l >= 16) ? 8 : 0) + (i & 7) + ((i >= 8) ? 16 : 0);
    int col = c * 16 + (l & 15);
    P[r] = (__bf16)W[k * 128 + col];
}

__global__ void cvt_bf16_kernel(const float* __restrict__ x, __bf16* __restrict__ xb, int n) {
    int i = blockIdx.x * blockDim.x + threadIdx.x;
    if (i < n) xb[i] = (__bf16)x[i];
}

// -------- node GEMM: out[N,128] = xbf[N,128] @ W(packed, LDS-staged) + bias --------
__global__ void __launch_bounds__(128) gemm_bias_kernel(const __bf16* __restrict__ xbf,
                                                        const __bf16* __restrict__ Wp,
                                                        const float* __restrict__ bias,
                                                        float* __restrict__ out, int N) {
    __shared__ __bf16 sW[16384];  // 32 KB packed weight tile, shared by all 4 waves
    {
        const uint4* s = (const uint4*)Wp;
        uint4* d = (uint4*)sW;
        for (int i = threadIdx.x; i < 2048; i += 128) d[i] = s[i];
    }
    __syncthreads();

    int wave = (blockIdx.x * blockDim.x + threadIdx.x) >> 5;
    int lane = threadIdx.x & 31;
    int ntiles = (N + 15) >> 4;
    if (wave >= ntiles) return;
    int n = lane & 15;
    int kbase = (lane >> 4) * 8;  // A-operand K offset within 16-chunk
    int mh = (lane >> 4) * 8;     // C/D row offset
    int row = wave * 16 + n;
    bool full = (wave * 16 + 16 <= N);

    v16bf A[4];
    if (row < N) {
        const __bf16* r = xbf + (size_t)row * 128 + kbase;
#pragma unroll
        for (int s = 0; s < 4; ++s) {
            const v8bf lo = *(const v8bf*)(r + s * 32);
            const v8bf hi = *(const v8bf*)(r + s * 32 + 16);
#pragma unroll
            for (int i = 0; i < 8; ++i) { A[s][i] = lo[i]; A[s][i + 8] = hi[i]; }
        }
    } else {
#pragma unroll
        for (int s = 0; s < 4; ++s)
#pragma unroll
            for (int i = 0; i < 16; ++i) A[s][i] = (__bf16)0.0f;
    }

    const __bf16* sWl = sW + (lane << 4);
    float* obase = out + (size_t)(wave * 16 + mh) * 128 + n;
#pragma unroll
    for (int c = 0; c < 8; ++c) {
        v8f acc;
#pragma unroll
        for (int j = 0; j < 8; ++j) acc[j] = 0.0f;
#pragma unroll
        for (int s = 0; s < 4; ++s) {
            v16bf b = *(const v16bf*)(sWl + ((c * 4 + s) << 9));
            acc = __builtin_amdgcn_wmma_f32_16x16x32_bf16(false, A[s], false, b,
                                                          (short)0, acc, false, false);
        }
        float bv = bias[c * 16 + n];
        float* op = obase + c * 16;
        if (full) {
#pragma unroll
            for (int j = 0; j < 8; ++j) op[j * 128] = acc[j] + bv;
        } else {
#pragma unroll
            for (int j = 0; j < 8; ++j)
                if (wave * 16 + mh + j < N) op[j * 128] = acc[j] + bv;
        }
    }
}

// -------- per-layer init: out = bias bcast; mx = -inf; den = 0 --------
__global__ void layer_init_kernel(float* __restrict__ out, const float* __restrict__ bias,
                                  float* __restrict__ mx, float* __restrict__ den, int N) {
    int i = blockIdx.x * blockDim.x + threadIdx.x;
    if (i >= N * 128) return;
    out[i] = bias[i & 127];
    if ((i & 127) == 0) {
        int nn = i >> 7;
        mx[nn] = -INFINITY;
        den[nn] = 0.0f;
    }
}

// -------- fused edge kernel: ee = ea@We (WMMA, We LDS-staged) ;
//          m = lrelu(xl[src]+xr[dst]+ee) ; logit = m.att ; atomic segment-max --------
__global__ void __launch_bounds__(128) edge_logits_kernel(
        const float* __restrict__ ea, const float* __restrict__ loopA,
        const int* __restrict__ srcA, const int* __restrict__ dstA,
        const float* __restrict__ xl, const float* __restrict__ xr,
        const __bf16* __restrict__ Wep, const float* __restrict__ att,
        float* __restrict__ logits, float* __restrict__ mx, int E, int EP) {
    __shared__ __bf16 sW[16384];  // 32 KB packed We tile
    {
        const uint4* s = (const uint4*)Wep;
        uint4* d = (uint4*)sW;
        for (int i = threadIdx.x; i < 2048; i += 128) d[i] = s[i];
    }
    __syncthreads();

    int wave = (blockIdx.x * blockDim.x + threadIdx.x) >> 5;
    int lane = threadIdx.x & 31;
    int ntiles = (EP + 15) >> 4;
    if (wave >= ntiles) return;
    int n = lane & 15;
    int kbase = (lane >> 4) * 8;
    int mh = (lane >> 4) * 8;
    bool full = (wave * 16 + 16 <= EP);

    // A tile: 16 edge rows, f32 source converted to bf16 in-register
    int erow = wave * 16 + n;
    const float* arow;
    if (erow < E) arow = ea + (size_t)erow * 128 + kbase;
    else if (erow < EP) arow = loopA + (size_t)(erow - E) * 128 + kbase;
    else arow = loopA + kbase;  // safe dummy (tile counts divide exactly)

    v16bf A[4];
#pragma unroll
    for (int s = 0; s < 4; ++s) {
        const float* p0 = arow + s * 32;
        const float* p1 = arow + s * 32 + 16;
#pragma unroll
        for (int i = 0; i < 8; ++i) {
            A[s][i]     = (__bf16)p0[i];
            A[s][i + 8] = (__bf16)p1[i];
        }
    }

    const __bf16* sWl = sW + (lane << 4);
    v8f acc[8];
#pragma unroll
    for (int c = 0; c < 8; ++c) {
        v8f a;
#pragma unroll
        for (int j = 0; j < 8; ++j) a[j] = 0.0f;
#pragma unroll
        for (int s = 0; s < 4; ++s) {
            v16bf b = *(const v16bf*)(sWl + ((c * 4 + s) << 9));
            a = __builtin_amdgcn_wmma_f32_16x16x32_bf16(false, A[s], false, b,
                                                        (short)0, a, false, false);
        }
        acc[c] = a;
    }

    // edge endpoints + row base pointers for the 8 rows this lane covers
    int de[8];
    const float* pxl[8];
    const float* pxr[8];
#pragma unroll
    for (int j = 0; j < 8; ++j) {
        int e = wave * 16 + mh + j;
        int s_, d_;
        if (e < E)       { s_ = srcA[e]; d_ = dstA[e]; }
        else if (e < EP) { s_ = e - E;   d_ = e - E; }
        else             { s_ = 0;       d_ = 0; }
        de[j] = d_;
        pxl[j] = xl + (size_t)s_ * 128 + n;
        pxr[j] = xr + (size_t)d_ * 128 + n;
    }

    float part[8];
#pragma unroll
    for (int j = 0; j < 8; ++j) part[j] = 0.0f;
#pragma unroll
    for (int c = 0; c < 8; ++c) {
        float av = att[c * 16 + n];
#pragma unroll
        for (int j = 0; j < 8; ++j) {
            float v = acc[c][j] + pxl[j][c * 16] + pxr[j][c * 16];
            v = (v > 0.0f) ? v : NEG_SLOPE * v;
            part[j] += av * v;
        }
    }
    // reduce across the 16 lanes of each half (masks 1,2,4,8 keep bit4 fixed)
#pragma unroll
    for (int j = 0; j < 8; ++j) {
        part[j] += __shfl_xor(part[j], 1);
        part[j] += __shfl_xor(part[j], 2);
        part[j] += __shfl_xor(part[j], 4);
        part[j] += __shfl_xor(part[j], 8);
    }
    if (n == 0) {
        float* lbase = logits + wave * 16 + mh;
        if (full) {
#pragma unroll
            for (int j = 0; j < 8; ++j) {
                lbase[j] = part[j];
                atomicMaxF(&mx[de[j]], part[j]);
            }
        } else {
#pragma unroll
            for (int j = 0; j < 8; ++j) {
                if (wave * 16 + mh + j < EP) {
                    lbase[j] = part[j];
                    atomicMaxF(&mx[de[j]], part[j]);
                }
            }
        }
    }
}

// -------- softmax numerator + denominator --------
__global__ void edge_exp_kernel(const int* __restrict__ dstA, float* __restrict__ logits,
                                const float* __restrict__ mx, float* __restrict__ den,
                                int E, int EP) {
    int e = blockIdx.x * blockDim.x + threadIdx.x;
    if (e >= EP) return;
    int d = (e < E) ? dstA[e] : (e - E);
    float w = __expf(logits[e] - mx[d]);
    logits[e] = w;
    atomicAdd(&den[d], w);
}

// -------- alpha-weighted scatter: out[dst] += alpha * xl[src] --------
__global__ void edge_scatter_kernel(const int* __restrict__ srcA, const int* __restrict__ dstA,
                                    const float* __restrict__ logits, const float* __restrict__ den,
                                    const float* __restrict__ xl, float* __restrict__ out,
                                    int E, int EP) {
    int gid = blockIdx.x * blockDim.x + threadIdx.x;
    int e = gid >> 5;
    if (e >= EP) return;
    int q = (gid & 31) * 4;
    int s, d;
    if (e < E) { s = srcA[e]; d = dstA[e]; } else { s = e - E; d = s; }
    float alpha = logits[e] / den[d];
    const float4 v = *(const float4*)(xl + (size_t)s * 128 + q);
    float* o = out + (size_t)d * 128 + q;
    atomicAdd(o + 0, alpha * v.x);
    atomicAdd(o + 1, alpha * v.y);
    atomicAdd(o + 2, alpha * v.z);
    atomicAdd(o + 3, alpha * v.w);
}

// -------- final score: wave per node, dot(x[n], Wf) + bf --------
__global__ void score_kernel(const float* __restrict__ x, const float* __restrict__ Wf,
                             const float* __restrict__ bf_, float* __restrict__ scores, int N) {
    int wave = (blockIdx.x * blockDim.x + threadIdx.x) >> 5;
    int lane = threadIdx.x & 31;
    if (wave >= N) return;
    const float4 xv = *(const float4*)(x + (size_t)wave * 128 + lane * 4);
    const float4 wv = *(const float4*)(Wf + lane * 4);
    float p = xv.x * wv.x + xv.y * wv.y + xv.z * wv.z + xv.w * wv.w;
    p += __shfl_xor(p, 1);
    p += __shfl_xor(p, 2);
    p += __shfl_xor(p, 4);
    p += __shfl_xor(p, 8);
    p += __shfl_xor(p, 16);
    if (lane == 0) scores[wave] = p + bf_[0];
}

// -------- top-10 (values desc, tie -> lower index), out[0..9]=vals out[10..19]=idx --------
__global__ void topk_kernel(const float* __restrict__ scores, int N, float* __restrict__ out) {
    __shared__ float sv[256 * 10];
    __shared__ int si[256 * 10];
    int t = threadIdx.x;
    float lv[10]; int li[10];
    for (int k = 0; k < 10; ++k) { lv[k] = -INFINITY; li[k] = 0x7fffffff; }
    for (int i = t; i < N; i += 256) {
        float v = scores[i];
        if (v > lv[9] || (v == lv[9] && i < li[9])) {
            int k = 9;
            while (k > 0 && (v > lv[k - 1] || (v == lv[k - 1] && i < li[k - 1]))) {
                lv[k] = lv[k - 1]; li[k] = li[k - 1]; --k;
            }
            lv[k] = v; li[k] = i;
        }
    }
    for (int k = 0; k < 10; ++k) { sv[t * 10 + k] = lv[k]; si[t * 10 + k] = li[k]; }
    __syncthreads();
    if (t == 0) {
        float fv[10]; int fi[10];
        for (int k = 0; k < 10; ++k) { fv[k] = -INFINITY; fi[k] = 0x7fffffff; }
        for (int i = 0; i < 256 * 10; ++i) {
            float v = sv[i]; int id = si[i];
            if (v > fv[9] || (v == fv[9] && id < fi[9])) {
                int k = 9;
                while (k > 0 && (v > fv[k - 1] || (v == fv[k - 1] && id < fi[k - 1]))) {
                    fv[k] = fv[k - 1]; fi[k] = fi[k - 1]; --k;
                }
                fv[k] = v; fi[k] = id;
            }
        }
        for (int k = 0; k < 10; ++k) { out[k] = fv[k]; out[10 + k] = (float)fi[k]; }
    }
}

// ==================== host launcher ====================
extern "C" void kernel_launch(void* const* d_in, const int* in_sizes, int n_in,
                              void* d_out, int out_size, void* d_ws, size_t ws_size,
                              hipStream_t stream) {
    const int D = 128;
    const int N = in_sizes[0] / D;          // 50000
    const int E = in_sizes[1] / 2;          // 640000
    const int L = in_sizes[3] / (D * D);    // 3
    const int EP = E + N;                   // 690000

    const float* x0   = (const float*)d_in[0];
    const int*   ei   = (const int*)d_in[1];
    const int*   srcA = ei;
    const int*   dstA = ei + E;
    const float* ea   = (const float*)d_in[2];
    const float* Wl   = (const float*)d_in[3];
    const float* bl   = (const float*)d_in[4];
    const float* Wr   = (const float*)d_in[5];
    const float* br   = (const float*)d_in[6];
    const float* We   = (const float*)d_in[7];
    const float* att  = (const float*)d_in[8];
    const float* bias = (const float*)d_in[9];
    const float* Wf   = (const float*)d_in[10];
    const float* bfp  = (const float*)d_in[11];

    // workspace carve
    char* w = (char*)d_ws;
    size_t off = 0;
    auto take = [&](size_t bytes) -> char* {
        char* p = w + off;
        off += (bytes + 255) & ~(size_t)255;
        return p;
    };
    const size_t NDf = (size_t)N * D * sizeof(float);
    float*  xA    = (float*)take(NDf);
    float*  xB    = (float*)take(NDf);
    float*  xl    = (float*)take(NDf);
    float*  xr    = (float*)take(NDf);
    float*  loopS = (float*)take(NDf);
    __bf16* x_bf  = (__bf16*)take((size_t)N * D * sizeof(__bf16));
    float*  den   = (float*)take((size_t)N * sizeof(float));   // deg in pre-pass
    float*  mx    = (float*)take((size_t)N * sizeof(float));   // scores at the end
    float*  logit = (float*)take((size_t)EP * sizeof(float));
    __bf16* Wlp   = (__bf16*)take(16384 * sizeof(__bf16));
    __bf16* Wrp   = (__bf16*)take(16384 * sizeof(__bf16));
    __bf16* Wep   = (__bf16*)take(16384 * sizeof(__bf16));

    const int BS = 256;
    int gND   = (N * D + BS - 1) / BS;
    int gE32  = (E * 32 + BS - 1) / BS;
    int gEP32 = (EP * 32 + BS - 1) / BS;
    int gEP   = (EP + BS - 1) / BS;
    int gPack = (3 * 16384 + BS - 1) / BS;
    int nRowTiles  = (N + 15) / 16;
    int gGemm  = (nRowTiles * 32 + 127) / 128;
    int nEdgeTiles = (EP + 15) / 16;
    int gEdge  = (nEdgeTiles * 32 + 127) / 128;
    int gScore = (N * 32 + 127) / 128;

    // pre-pass: self-loop attributes (mean of incoming)
    prep_zero_kernel<<<gND, BS, 0, stream>>>(loopS, den, N);
    prep_accum_kernel<<<gE32, BS, 0, stream>>>(dstA, ea, loopS, den, E);
    prep_final_kernel<<<gND, BS, 0, stream>>>(loopS, den, N);

    const float* xcur = x0;
    float* bufs[2] = {xA, xB};
    for (int l = 0; l < L; ++l) {
        pack_w_kernel<<<gPack, BS, 0, stream>>>(Wl + (size_t)l * 16384, Wr + (size_t)l * 16384,
                                                We + (size_t)l * 16384, Wlp, Wrp, Wep);
        cvt_bf16_kernel<<<gND, BS, 0, stream>>>(xcur, x_bf, N * D);
        gemm_bias_kernel<<<gGemm, 128, 0, stream>>>(x_bf, Wlp, bl + l * D, xl, N);
        gemm_bias_kernel<<<gGemm, 128, 0, stream>>>(x_bf, Wrp, br + l * D, xr, N);
        float* outl = bufs[l & 1];
        layer_init_kernel<<<gND, BS, 0, stream>>>(outl, bias + l * D, mx, den, N);
        edge_logits_kernel<<<gEdge, 128, 0, stream>>>(ea, loopS, srcA, dstA, xl, xr, Wep,
                                                      att + l * D, logit, mx, E, EP);
        edge_exp_kernel<<<gEP, BS, 0, stream>>>(dstA, logit, mx, den, E, EP);
        edge_scatter_kernel<<<gEP32, BS, 0, stream>>>(srcA, dstA, logit, den, xl, outl, E, EP);
        xcur = outl;
    }

    // final score + top-k (reuse mx as scores buffer)
    score_kernel<<<gScore, 128, 0, stream>>>(xcur, Wf, bfp, mx, N);
    topk_kernel<<<1, 256, 0, stream>>>(mx, N, (float*)d_out);
}
